// MDSA2_76690936038056
// MI455X (gfx1250) — matrix operand order repaired and checked by reading, not compiled, not run on hardware
//
#include <hip/hip_runtime.h>
#include <hip/hip_bf16.h>

// ---------------------------------------------------------------------------
// MDSA axial attention for MI455X (gfx1250).
// B=1, C=32, D=32, H=128, W=128.
//   - QKV / Wo projections: V_WMMA_F32_16X16X4_F32 (exact f32 math)
//   - Axial attention:      V_WMMA_F32_16X16X32_BF16 (bf16 staged q/k/v, K=C=32)
//   - Softmax reductions:   v_permlane16_b32 (VALU, keeps LDS pipe free for frags)
//   - D-axis kernel stores o_acc (no zero-fill / RMW needed); H/W accumulate.
// ---------------------------------------------------------------------------

#define CCH   32
#define DD    32
#define HH    128
#define WW    128
#define NTOT  (DD * HH * WW)            // 524288 voxels

typedef __attribute__((ext_vector_type(16))) __bf16 v16bf;
typedef __attribute__((ext_vector_type(8)))  float  v8f;
typedef __attribute__((ext_vector_type(2)))  float  v2f;

// float -> bf16 bits, round-to-nearest-even
__device__ __forceinline__ unsigned short f2bf(float f) {
    unsigned int u = __float_as_uint(f);
    u += 0x7FFFu + ((u >> 16) & 1u);
    return (unsigned short)(u >> 16);
}

// ---- cross-lane xor reductions within 16-lane rows (VALU permlane path) ----
#if __has_builtin(__builtin_amdgcn_permlane16)
__device__ __forceinline__ float pl16(float v, int lo, int hi) {
    return __int_as_float(
        __builtin_amdgcn_permlane16(__float_as_int(v), __float_as_int(v), lo, hi, false, false));
}
__device__ __forceinline__ float red16_max(float v) {
    v = fmaxf(v, pl16(v, 0x67452301, (int)0xefcdab89));  // xor 1
    v = fmaxf(v, pl16(v, 0x54761032, (int)0xdcfeba98));  // xor 2 (lanes 8-11 of each row select themselves)
    v = fmaxf(v, pl16(v, 0x32107654, (int)0xba98fedc));  // xor 4
    v = fmaxf(v, pl16(v, (int)0xfedcba98, 0x76543210));  // xor 8
    return v;
}
__device__ __forceinline__ float red16_sum(float v) {
    v += pl16(v, 0x67452301, (int)0xefcdab89);
    v += pl16(v, 0x54761032, (int)0xdcfeba98);
    v += pl16(v, 0x32107654, (int)0xba98fedc);
    v += pl16(v, (int)0xfedcba98, 0x76543210);
    return v;
}
#else
__device__ __forceinline__ float red16_max(float v) {
    for (int off = 1; off < 16; off <<= 1) v = fmaxf(v, __shfl_xor(v, off, 32));
    return v;
}
__device__ __forceinline__ float red16_sum(float v) {
    for (int off = 1; off < 16; off <<= 1) v += __shfl_xor(v, off, 32);
    return v;
}
#endif

// A-matrix fragment, 16x32 bf16 (M x K), LDS row-major [m][k], pitch in halves.
__device__ __forceinline__ v16bf load_frag_a(const unsigned short* lds, int base, int pitch) {
    const int lane = threadIdx.x & 31;
    const int m  = lane & 15;
    const int kb = (lane < 16) ? 0 : 8;
    const unsigned short* p = lds + base + m * pitch + kb;
    union { uint4 u[2]; v16bf v; } t;
    t.u[0] = *(const uint4*)(p);        // K = kb .. kb+7
    t.u[1] = *(const uint4*)(p + 16);   // K = kb+16 .. kb+23
    return t.v;
}

// B-matrix fragment, 32x16 bf16 (K x N), LDS stored as [n][k] row-major.
__device__ __forceinline__ v16bf load_frag_b(const unsigned short* lds, int base, int pitch) {
    const int lane = threadIdx.x & 31;
    const int n  = lane & 15;
    const int kb = (lane < 16) ? 0 : 8;
    const unsigned short* p = lds + base + n * pitch + kb;
    union { uint4 u[2]; v16bf v; } t;
    t.u[0] = *(const uint4*)(p);
    t.u[1] = *(const uint4*)(p + 16);
    return t.v;
}

__device__ __forceinline__ v8f wmma_bf16(v16bf a, v16bf b, v8f c) {
    return __builtin_amdgcn_wmma_f32_16x16x32_bf16(false, a, false, b, (short)0, c, false, false);
}
__device__ __forceinline__ v8f wmma_f32(v2f a, v2f b, v8f c) {
    return __builtin_amdgcn_wmma_f32_16x16x4_f32(false, a, false, b, (short)0, c, false, false);
}

// ---------------------------------------------------------------------------
// Kernel 1: q/k/v = W{q,k,v} * x via f32 WMMA (exact), bf16 store.
// 256 threads = 8 waves; each wave owns a 16-voxel tile (128 voxels / block).
// ---------------------------------------------------------------------------
__global__ __launch_bounds__(256) void qkv_proj_kernel(
    const float* __restrict__ x,
    const float* __restrict__ Wq, const float* __restrict__ Wk, const float* __restrict__ Wv,
    unsigned short* __restrict__ qg, unsigned short* __restrict__ kg,
    unsigned short* __restrict__ vg)
{
    __shared__ float w[3][CCH][CCH];
    const int tid = threadIdx.x;
    for (int i = tid; i < CCH * CCH; i += 256) {
        w[0][0][i] = Wq[i];
        w[1][0][i] = Wk[i];
        w[2][0][i] = Wv[i];
    }
    __syncthreads();

    const int wave = tid >> 5, lane = tid & 31;
    const int n    = lane & 15;                  // voxel within tile / A-row
    const int koff = (lane < 16) ? 0 : 2;        // f32 WMMA: K = koff + e (A and B)
    const int hsel = (lane < 16) ? 0 : 8;        // C/D frag: M = r + hsel
    const size_t v0 = (size_t)blockIdx.x * 128 + (size_t)wave * 16;

    const v8f zero = {0.f, 0.f, 0.f, 0.f, 0.f, 0.f, 0.f, 0.f};
    v8f acc[3][2];
#pragma unroll
    for (int mat = 0; mat < 3; ++mat) { acc[mat][0] = zero; acc[mat][1] = zero; }

#pragma unroll
    for (int ks = 0; ks < 8; ++ks) {             // K = 32 channels, 4 per step
        const int c0 = 4 * ks + koff;
        v2f b;
        b.x = x[(size_t)c0 * NTOT + v0 + n];     // B frag straight from [c][v] layout
        b.y = x[(size_t)(c0 + 1) * NTOT + v0 + n];
#pragma unroll
        for (int mat = 0; mat < 3; ++mat)
#pragma unroll
            for (int mt = 0; mt < 2; ++mt) {     // output channel tiles 0-15 / 16-31
                v2f a;
                a.x = w[mat][mt * 16 + n][c0];
                a.y = w[mat][mt * 16 + n][c0 + 1];
                acc[mat][mt] = wmma_f32(a, b, acc[mat][mt]);
            }
    }

#pragma unroll
    for (int mt = 0; mt < 2; ++mt)
#pragma unroll
        for (int r = 0; r < 8; ++r) {
            const int o = mt * 16 + r + hsel;
            const size_t g = (size_t)o * NTOT + v0 + n;
            qg[g] = f2bf(acc[0][mt][r]);
            kg[g] = f2bf(acc[1][mt][r]);
            vg[g] = f2bf(acc[2][mt][r]);
        }
}

// ---------------------------------------------------------------------------
// Kernel 2/3/4: axial attention along one axis (length L), bf16 WMMA.
//   AXIS 0: D (stride 16384) ; AXIS 1: H (stride 128) ; AXIS 2: W (stride 1)
// ACCUM=false: store o_acc (first axis); ACCUM=true: accumulate.
// ---------------------------------------------------------------------------
template <int L, int AXIS, bool ACCUM>
__global__ __launch_bounds__(256) void attn_axis_kernel(
    const unsigned short* __restrict__ qg, const unsigned short* __restrict__ kg,
    const unsigned short* __restrict__ vg, float* __restrict__ o_acc)
{
    constexpr int WPL = L / 16;          // waves per line
    constexpr int LPB = 8 / WPL;         // lines per block
    constexpr int NT  = L / 16;          // 16-col tiles per score row
    constexpr int QKP = 40;              // Q/K LDS pitch (halves)
    constexpr int VP  = L + 8;
    constexpr int PP  = L + 8;
    constexpr int QOFF = 0;
    constexpr int KOFF = L * QKP;
    constexpr int VOFF = 2 * L * QKP;
    constexpr int POFF = VOFF + CCH * VP;
    constexpr int LINE_HALVES = POFF + L * PP;
    constexpr int LINE_STRIDE = LINE_HALVES + 8;   // de-bank-align line regions (16B-aligned)

    __shared__ unsigned short smem[LINE_STRIDE * LPB];

    const int tid = threadIdx.x;
    const float scale = 0.17677669529663687f;    // 1/sqrt(C)

    // ---- stage q[pos][c], k[pos][c], v[c][pos] for LPB lines ----
    // Line index varies fastest: for AXIS 0 the LPB=4 lines sit at consecutive
    // w, so adjacent lanes read contiguous bf16 despite the 16384-elem stride.
    for (int idx = tid; idx < LPB * CCH * L; idx += 256) {
        const int li  = idx % LPB;
        const int t   = idx / LPB;
        const int c   = t / L;
        const int pos = t - c * L;
        const int l   = blockIdx.x * LPB + li;
        int base, stride;
        if (AXIS == 0)      { stride = HH * WW; base = l; }
        else if (AXIS == 1) { stride = WW;      base = (l >> 7) * (HH * WW) + (l & (WW - 1)); }
        else                { stride = 1;       base = l * WW; }
        const size_t g = (size_t)c * NTOT + base + (size_t)pos * stride;
        unsigned short* ls = smem + li * LINE_STRIDE;
        ls[QOFF + pos * QKP + c] = qg[g];
        ls[KOFF + pos * QKP + c] = kg[g];
        ls[VOFF + c * VP + pos]  = vg[g];
    }
    __syncthreads();

    const int wave = tid >> 5;
    const int lane = tid & 31;
    const int li   = wave / WPL;
    const int wl   = wave % WPL;
    const int m0   = wl * 16;
    const int lb   = li * LINE_STRIDE;
    const int n    = lane & 15;
    const int hsel = (lane < 16) ? 0 : 8;

    // ---- scores S = Q^T K : one wmma per 16x16 tile (K-dim = C = 32) ----
    v8f acc[NT];
    {
        const v16bf afrag = load_frag_a(smem, lb + QOFF + m0 * QKP, QKP);
        const v8f zero = {0.f, 0.f, 0.f, 0.f, 0.f, 0.f, 0.f, 0.f};
#pragma unroll
        for (int j = 0; j < NT; ++j) {
            const v16bf bfrag = load_frag_b(smem, lb + KOFF + j * 16 * QKP, QKP);
            acc[j] = wmma_bf16(afrag, bfrag, zero);
        }
    }

    // ---- row softmax (row spread across a 16-lane half) ----
#pragma unroll
    for (int r = 0; r < 8; ++r) {
        float mx = acc[0][r];
#pragma unroll
        for (int j = 1; j < NT; ++j) mx = fmaxf(mx, acc[j][r]);
        mx = red16_max(mx);

        float sum = 0.f;
#pragma unroll
        for (int j = 0; j < NT; ++j) {
            const float e = __expf((acc[j][r] - mx) * scale);
            acc[j][r] = e;
            sum += e;
        }
        sum = red16_sum(sum);
        const float inv = 1.0f / sum;

        const int row = m0 + r + hsel;
#pragma unroll
        for (int j = 0; j < NT; ++j)
            smem[lb + POFF + row * PP + j * 16 + n] = f2bf(acc[j][r] * inv);
    }
    __syncthreads();

    // ---- O = V * A^T : A-frag from V[c][u], B-frag from P[w][u] ----
    {
        const int j = wl;
        const int l = blockIdx.x * LPB + li;
        int base, stride;
        if (AXIS == 0)      { stride = HH * WW; base = l; }
        else if (AXIS == 1) { stride = WW;      base = (l >> 7) * (HH * WW) + (l & (WW - 1)); }
        else                { stride = 1;       base = l * WW; }

#pragma unroll
        for (int mt = 0; mt < 2; ++mt) {
            v8f o = {0.f, 0.f, 0.f, 0.f, 0.f, 0.f, 0.f, 0.f};
#pragma unroll
            for (int ks = 0; ks < L / 32; ++ks) {
                const v16bf a = load_frag_a(smem, lb + VOFF + mt * 16 * VP + ks * 32, VP);
                const v16bf b = load_frag_b(smem, lb + POFF + j * 16 * PP + ks * 32, PP);
                o = wmma_bf16(a, b, o);
            }
#pragma unroll
            for (int r = 0; r < 8; ++r) {
                const int c   = mt * 16 + r + hsel;
                const int pos = j * 16 + n;
                const size_t g = (size_t)c * NTOT + base + (size_t)pos * stride;
                if (ACCUM) o_acc[g] += o[r];     // single writer per element per kernel
                else       o_acc[g]  = o[r];
            }
        }
    }
}

// ---------------------------------------------------------------------------
// Kernel 5: out = Wo * (od + oh + ow) + x via f32 WMMA (exact).
// ---------------------------------------------------------------------------
__global__ __launch_bounds__(256) void out_proj_kernel(
    const float* __restrict__ x, const float* __restrict__ Wo,
    const float* __restrict__ o_acc, float* __restrict__ out)
{
    __shared__ float w[CCH][CCH];
    const int tid = threadIdx.x;
    for (int i = tid; i < CCH * CCH; i += 256) w[0][i] = Wo[i];
    __syncthreads();

    const int wave = tid >> 5, lane = tid & 31;
    const int n    = lane & 15;
    const int koff = (lane < 16) ? 0 : 2;
    const int hsel = (lane < 16) ? 0 : 8;
    const size_t v0 = (size_t)blockIdx.x * 128 + (size_t)wave * 16;

    const v8f zero = {0.f, 0.f, 0.f, 0.f, 0.f, 0.f, 0.f, 0.f};
    v8f acc[2] = {zero, zero};

#pragma unroll
    for (int ks = 0; ks < 8; ++ks) {
        const int c0 = 4 * ks + koff;
        v2f b;
        b.x = o_acc[(size_t)c0 * NTOT + v0 + n];
        b.y = o_acc[(size_t)(c0 + 1) * NTOT + v0 + n];
#pragma unroll
        for (int mt = 0; mt < 2; ++mt) {
            v2f a;
            a.x = w[mt * 16 + n][c0];
            a.y = w[mt * 16 + n][c0 + 1];
            acc[mt] = wmma_f32(a, b, acc[mt]);
        }
    }

#pragma unroll
    for (int mt = 0; mt < 2; ++mt)
#pragma unroll
        for (int r = 0; r < 8; ++r) {
            const int o = mt * 16 + r + hsel;
            const size_t g = (size_t)o * NTOT + v0 + n;
            out[g] = acc[mt][r] + x[g];          // residual
        }
}

// ---------------------------------------------------------------------------
extern "C" void kernel_launch(void* const* d_in, const int* in_sizes, int n_in,
                              void* d_out, int out_size, void* d_ws, size_t ws_size,
                              hipStream_t stream) {
    const float* x  = (const float*)d_in[0];
    const float* Wq = (const float*)d_in[1];
    const float* Wk = (const float*)d_in[2];
    const float* Wv = (const float*)d_in[3];
    const float* Wo = (const float*)d_in[4];
    float* out = (float*)d_out;

    const size_t bf16_plane = (size_t)CCH * NTOT;
    unsigned short* qg = (unsigned short*)d_ws;
    unsigned short* kg = qg + bf16_plane;
    unsigned short* vg = kg + bf16_plane;
    float* o_acc = (float*)(vg + bf16_plane);

    const int tile_blocks = NTOT / 128;                      // 4096

    qkv_proj_kernel<<<tile_blocks, 256, 0, stream>>>(x, Wq, Wk, Wv, qg, kg, vg);

    attn_axis_kernel<32, 0, false><<<(HH * WW) / 4, 256, 0, stream>>>(qg, kg, vg, o_acc);
    attn_axis_kernel<128, 1, true><<<DD * WW, 256, 0, stream>>>(qg, kg, vg, o_acc);
    attn_axis_kernel<128, 2, true><<<DD * HH, 256, 0, stream>>>(qg, kg, vg, o_acc);

    out_proj_kernel<<<tile_blocks, 256, 0, stream>>>(x, Wo, o_acc, out);
}
